// UniversalSAE_14577119002707
// MI455X (gfx1250) — compile-verified, matrix operand order checked by reading, not run on hardware
//
#include <hip/hip_runtime.h>

// ---------------------------------------------------------------------------
// Universal SAE forward for MI455X (gfx1250, wave32).
//   z_dense = (x - pre_bias) @ enc_w^T   (bf16 WMMA, f32 accumulate,
//                                         async-to-LDS double-buffered B tile)
//   z       = TopK32(z_dense)            (LDS iterative argmax)
//   rec_i   = z_sparse @ dec_w_i^T + post_bias_i   (sparse gather decode)
//
// d_ws layout (assumes ws_size >= 64 MB):
//   [0, 32MB)   : bf16 A' = bf16(x_sel - pre_bias_sel), 16384 x dim
//   [32, 48MB)  : bf16 W  = bf16(enc_w_sel),            8192  x dim
//   [48, 52MB)  : (float val, int idx) pairs, 16384 x 32
// ---------------------------------------------------------------------------

#define SAE_B   16384
#define SAE_H   8192
#define SAE_D0  512
#define SAE_D1  768

typedef __attribute__((ext_vector_type(16))) __bf16     v16bf;
typedef __attribute__((ext_vector_type(8)))  float      v8f;
typedef __attribute__((ext_vector_type(4)))  unsigned   u32x4;

// ---- float -> bf16 round-to-nearest-even ----------------------------------
__device__ __forceinline__ unsigned short f32_to_bf16_rne(float f) {
    unsigned u = __float_as_uint(f);
    unsigned r = u + 0x7fffu + ((u >> 16) & 1u);
    return (unsigned short)(r >> 16);
}

// ---- CDNA5 async copy: global -> LDS, 16B per lane, ASYNCcnt-tracked ------
__device__ __forceinline__ void async_copy_b128(unsigned lds_off,
                                                const void* gaddr) {
    asm volatile("global_load_async_to_lds_b128 %0, %1, off"
                 :: "v"(lds_off), "v"(gaddr) : "memory");
}
template <int N>
__device__ __forceinline__ void wait_asynccnt_le() {
    asm volatile("s_wait_asynccnt %0" :: "i"(N) : "memory");
}

// ---- kernel 1: A' = bf16(x_sel - pre_bias_sel) ----------------------------
__global__ void sae_cvt_x(const float* __restrict__ x0, const float* __restrict__ x1,
                          const float* __restrict__ pb0, const float* __restrict__ pb1,
                          const int* __restrict__ src, unsigned short* __restrict__ out) {
    const int s   = *src;
    const int dim = s ? SAE_D1 : SAE_D0;
    const float* __restrict__ x  = s ? x1  : x0;
    const float* __restrict__ pb = s ? pb1 : pb0;
    const int row = blockIdx.x;                       // 16384 blocks
    const float* xr = x + (size_t)row * dim;
    unsigned short* orow = out + (size_t)row * dim;
    for (int k = threadIdx.x; k < dim; k += blockDim.x)
        orow[k] = f32_to_bf16_rne(xr[k] - pb[k]);
}

// ---- kernel 2: W = bf16(enc_w_sel) ----------------------------------------
__global__ void sae_cvt_w(const float* __restrict__ w0, const float* __restrict__ w1,
                          const int* __restrict__ src, unsigned short* __restrict__ out) {
    const int s   = *src;
    const int dim = s ? SAE_D1 : SAE_D0;
    const float* __restrict__ w = s ? w1 : w0;
    const int row = blockIdx.x;                       // 8192 blocks
    const float* wr = w + (size_t)row * dim;
    unsigned short* orow = out + (size_t)row * dim;
    for (int k = threadIdx.x; k < dim; k += blockDim.x)
        orow[k] = f32_to_bf16_rne(wr[k]);
}

// ---- fragment helpers matching CDNA5 16-bit A/B WMMA VGPR layout ----------
// Lanes 0-15 hold row/col (lane), K = base+0..7 and base+16..23.
// Lanes 16-31 hold row/col (lane-16), K = base+8..15 and base+24..31.
__device__ __forceinline__ v16bf frag_from_ptr(const unsigned short* p) {
    union { u32x4 q[2]; v16bf v; } u;
    u.q[0] = *(const u32x4*)(p);        // 8 bf16: K + 0..7   (or +8..15)
    u.q[1] = *(const u32x4*)(p + 16);   // 8 bf16: K + 16..23 (or +24..31)
    return u.v;
}

// ---- kernel 3: encode GEMM, z_dense[16384 x 8192] -------------------------
// Block = 256 threads (8 waves). Block tile: 128 (M) x 128 (N).
// Wave w: M rows [blockM*128 + w*16, +16), all 128 N cols -> 8 WMMA tiles.
// B tile (128 n-rows x 32 k, bf16, 8KB) staged in LDS via async copy,
// double-buffered; A fragment double-buffered in registers.
__global__ void __launch_bounds__(256)
sae_encode_gemm(const unsigned short* __restrict__ A,   // bf16 [B x K]
                const unsigned short* __restrict__ W,   // bf16 [H x K]
                float* __restrict__ z,                  // f32  [B x H]
                const int* __restrict__ src) {
    __shared__ __align__(16) unsigned short btile[2][128 * 32];   // 2 x 8KB

    const int K      = (*src) ? SAE_D1 : SAE_D0;        // 512 or 768, both %32==0
    const int ktiles = K >> 5;

    const int tid  = threadIdx.x;
    const int wave = tid >> 5;
    const int lane = tid & 31;
    const int l15  = lane & 15;
    const int lhi  = lane >> 4;                         // 0: lanes 0-15, 1: 16-31

    const int m0 = blockIdx.y * 128 + wave * 16;        // 16-row strip per wave
    const int n0 = blockIdx.x * 128;

    // cooperative async stage of one 128x32 B tile into btile[buf]
    auto stage_b = [&](int kt, int buf) {
        const int k32 = kt << 5;
        #pragma unroll
        for (int j = 0; j < 2; ++j) {                   // 512 16B segments / 256 thr
            const int seg  = tid + j * 256;
            const int rowN = seg >> 2;                  // 0..127 (block-local n)
            const int part = seg & 3;                   // 16B chunk within 64B row
            const unsigned short* g =
                W + (size_t)(n0 + rowN) * K + k32 + part * 8;
            const unsigned loff =
                (unsigned)(uintptr_t)&btile[buf][rowN * 32 + part * 8];
            async_copy_b128(loff, g);
        }
    };

    v8f acc[8];
    #pragma unroll
    for (int j = 0; j < 8; ++j) acc[j] = (v8f)(0.0f);

    const int arow = m0 + l15;
    const unsigned short* aptr = A + (size_t)arow * K + (lhi << 3);

    stage_b(0, 0);                                      // prologue
    v16bf af_cur = frag_from_ptr(aptr);                 // A fragment, kt = 0

    for (int kt = 0; kt < ktiles; ++kt) {
        const bool more = (kt + 1) < ktiles;
        if (more) stage_b(kt + 1, (kt + 1) & 1);        // prefetch next B tile

        v16bf af_next;
        if (more) {
            __builtin_prefetch(aptr + ((kt + 1) << 5) + 32, 0, 1);
            af_next = frag_from_ptr(aptr + ((kt + 1) << 5));
        }

        // current tile's 2 async ops done (next tile's 2 may stay in flight)
        if (more) wait_asynccnt_le<2>(); else wait_asynccnt_le<0>();
        __syncthreads();                                // tile visible to all waves

        const unsigned short* bbase = &btile[kt & 1][lhi * 8];
        #pragma unroll
        for (int j = 0; j < 8; ++j) {
            const v16bf bf = frag_from_ptr(bbase + (j * 16 + l15) * 32);
            acc[j] = __builtin_amdgcn_wmma_f32_16x16x32_bf16(
                false, af_cur, false, bf, (short)0, acc[j], false, false);
        }
        __syncthreads();            // all waves done reading before overwrite
        af_cur = af_next;
    }

    // D layout: VGPR r, lanes 0-15 -> (M = r,   N = lane)
    //                   lanes 16-31 -> (M = r+8, N = lane-16)
    #pragma unroll
    for (int j = 0; j < 8; ++j) {
        union { v8f v; float f[8]; } u; u.v = acc[j];
        const int ncol = n0 + j * 16 + l15;
        #pragma unroll
        for (int r = 0; r < 8; ++r) {
            const int mrow = m0 + r + 8 * lhi;
            z[(size_t)mrow * SAE_H + ncol] = u.f[r];
        }
    }
}

// ---- kernel 4: per-row Top-32 sparsification ------------------------------
#define TK_TPB 256
__global__ void __launch_bounds__(TK_TPB)
sae_topk(float* __restrict__ z, float* __restrict__ pairs) {
    __shared__ float         vals[SAE_H];     // 32 KB
    __shared__ unsigned char flag[SAE_H];     // 8 KB
    __shared__ float         redV[TK_TPB];
    __shared__ int           redI[TK_TPB];
    __shared__ float         selV[32];
    __shared__ int           selI[32];

    const int row = blockIdx.x;
    const int tid = threadIdx.x;
    float* zr = z + (size_t)row * SAE_H;

    for (int i = tid; i < SAE_H; i += TK_TPB) { vals[i] = zr[i]; flag[i] = 0; }
    __syncthreads();

    for (int it = 0; it < 32; ++it) {
        float bv = -3.402823466e38f; int bi = -1;
        for (int i = tid; i < SAE_H; i += TK_TPB) {
            float v = vals[i];
            if (!flag[i] && (v > bv || (v == bv && (unsigned)i < (unsigned)bi))) {
                bv = v; bi = i;
            }
        }
        redV[tid] = bv; redI[tid] = bi;
        __syncthreads();
        #pragma unroll
        for (int s = TK_TPB / 2; s > 0; s >>= 1) {
            if (tid < s) {
                float ov = redV[tid + s]; int oi = redI[tid + s];
                if (ov > redV[tid] ||
                    (ov == redV[tid] && (unsigned)oi < (unsigned)redI[tid])) {
                    redV[tid] = ov; redI[tid] = oi;
                }
            }
            __syncthreads();
        }
        if (tid == 0) {
            int w = redI[0];
            selI[it] = w; selV[it] = redV[0];
            if (w >= 0) flag[w] = 1;
        }
        __syncthreads();
    }

    for (int i = tid; i < SAE_H; i += TK_TPB) zr[i] = flag[i] ? vals[i] : 0.0f;
    if (tid < 32) {
        float2 e; e.x = selV[tid]; e.y = __int_as_float(selI[tid]);
        ((float2*)pairs)[(size_t)row * 32 + tid] = e;
    }
}

// ---- kernel 5: sparse decode into both activation spaces ------------------
__global__ void __launch_bounds__(256)
sae_decode(const float* __restrict__ pairs,
           const float* __restrict__ dw0, const float* __restrict__ dw1,
           const float* __restrict__ pb0, const float* __restrict__ pb1,
           float* __restrict__ rec0, float* __restrict__ rec1) {
    __shared__ float sv[32];
    __shared__ int   si[32];
    const int row = blockIdx.x;
    if (threadIdx.x < 32) {
        float2 e = ((const float2*)pairs)[(size_t)row * 32 + threadIdx.x];
        sv[threadIdx.x] = e.x;
        si[threadIdx.x] = __float_as_int(e.y);
    }
    __syncthreads();

    for (int d = threadIdx.x; d < (SAE_D0 + SAE_D1); d += 256) {
        const bool is0 = d < SAE_D0;
        const int  dd  = is0 ? d : d - SAE_D0;
        const float* __restrict__ w = is0 ? (dw0 + (size_t)dd * SAE_H)
                                          : (dw1 + (size_t)dd * SAE_H);
        float acc = is0 ? pb0[dd] : pb1[dd];
        #pragma unroll
        for (int j = 0; j < 32; ++j) {
            int h = si[j];
            if (h >= 0) acc = fmaf(sv[j], w[h], acc);
        }
        if (is0) rec0[(size_t)row * SAE_D0 + dd] = acc;
        else     rec1[(size_t)row * SAE_D1 + dd] = acc;
    }
}

// ---------------------------------------------------------------------------
extern "C" void kernel_launch(void* const* d_in, const int* in_sizes, int n_in,
                              void* d_out, int out_size, void* d_ws, size_t ws_size,
                              hipStream_t stream) {
    (void)in_sizes; (void)n_in; (void)out_size; (void)ws_size;

    const float* x0  = (const float*)d_in[0];
    const float* x1  = (const float*)d_in[1];
    const float* pb0 = (const float*)d_in[2];
    const float* pb1 = (const float*)d_in[3];
    const float* ew0 = (const float*)d_in[4];
    const float* ew1 = (const float*)d_in[5];
    const float* dw0 = (const float*)d_in[6];
    const float* dw1 = (const float*)d_in[7];
    const float* ob0 = (const float*)d_in[8];
    const float* ob1 = (const float*)d_in[9];
    const int*   src = (const int*)d_in[10];

    float* z    = (float*)d_out;                          // [B x H]
    float* rec0 = z + (size_t)SAE_B * SAE_H;              // [B x D0]
    float* rec1 = rec0 + (size_t)SAE_B * SAE_D0;          // [B x D1]

    unsigned short* abf   = (unsigned short*)d_ws;
    unsigned short* wbf   = (unsigned short*)((char*)d_ws + (size_t)(32u << 20));
    float*          pairs = (float*)((char*)d_ws + (size_t)(48u << 20));

    sae_cvt_x<<<SAE_B, 256, 0, stream>>>(x0, x1, pb0, pb1, src, abf);
    sae_cvt_w<<<SAE_H, 256, 0, stream>>>(ew0, ew1, src, wbf);

    dim3 grid(SAE_H / 128, SAE_B / 128);                  // (64, 128)
    sae_encode_gemm<<<grid, 256, 0, stream>>>(abf, wbf, z, src);

    sae_topk<<<SAE_B, TK_TPB, 0, stream>>>(z, pairs);

    sae_decode<<<SAE_B, 256, 0, stream>>>(pairs, dw0, dw1, ob0, ob1, rec0, rec1);
}
